// VQVAE_46454366273916
// MI455X (gfx1250) — matrix operand order repaired and checked by reading, not compiled, not run on hardware
//
#include <hip/hip_runtime.h>

typedef __bf16 bf16;
typedef __attribute__((ext_vector_type(16))) __bf16 v16bf;
typedef __attribute__((ext_vector_type(8)))  __bf16 v8bf;   // 16 bytes
typedef __attribute__((ext_vector_type(8)))  float  v8f;

#define TBM 128
#define TBN 128
#define TBK 32
#define LDT 40   // padded LDS row length in bf16 (80B: 16B-aligned chunks, bank-spread)

union AFrag { v16bf v; v8bf h[2]; };
union CFrag { v8f  v; float e[8]; };

// ---------------------------------------------------------------------------
// WMMA implicit-GEMM: out(NCHW) = A[M][Kp] * B[N][Kp]^T (+bias[m]) (+res[idx])
// Double-buffered LDS staging via gfx1250 async global->LDS DMA
// (global_load_async_to_lds_b128, ASYNCcnt) overlapped with the WMMA stream.
// B (im2col) is stored n-major / k-contiguous so fragments are ds_load_b128
// pairs matching the ISA 7.12.2 wave32 layouts.
// n -> (b, yy, xx) over PH x PW phase grid; output pixel (yy*sy+py, xx*sx+px).
// ---------------------------------------------------------------------------
__global__ __launch_bounds__(256) void gemm_bf16_wmma_nchw(
    const bf16* __restrict__ A, const bf16* __restrict__ Bm,
    const float* __restrict__ bias, const float* __restrict__ res,
    float* __restrict__ out,
    int M, int N, int Kp,
    int Cout, int OH, int OW, int PH, int PW,
    int sy, int sx, int py, int px)
{
    __shared__ __align__(16) bf16 As[2][TBM][LDT];   // [stage][m][k]
    __shared__ __align__(16) bf16 Bs[2][TBN][LDT];   // [stage][n][k] (transposed tile)

    const int tid  = threadIdx.x;
    const int lane = tid & 31;
    const int wid  = tid >> 5;
    const int wm   = wid & 1;    // 2 waves along M (64 rows each)
    const int wn   = wid >> 1;   // 4 waves along N (32 cols each)
    const int m0   = blockIdx.y * TBM;
    const int n0   = blockIdx.x * TBN;

    CFrag acc[4][2];
#pragma unroll
    for (int i = 0; i < 4; ++i)
#pragma unroll
        for (int j = 0; j < 2; ++j)
#pragma unroll
            for (int r = 0; r < 8; ++r) acc[i][j].e[r] = 0.f;

    // staging: 2 threads per row, 16 bf16 (2 x 16B async DMAs) each
    const int srow = tid >> 1;
    const int scs  = (tid & 1) * 16;
    const int mRow = m0 + srow;
    const bool mOK = (mRow < M);

    // Guard rows (m >= M) are zeroed once in both stages and never overwritten
    // (async loads for them are simply not issued).
    if (!mOK) {
        const v8bf zero8 = {};
        *(v8bf*)&As[0][srow][scs]     = zero8;
        *(v8bf*)&As[0][srow][scs + 8] = zero8;
        *(v8bf*)&As[1][srow][scs]     = zero8;
        *(v8bf*)&As[1][srow][scs + 8] = zero8;
    }

    const bf16* gA = A  + (size_t)mRow * Kp + scs;         // + kt at issue time
    const bf16* gB = Bm + (size_t)(n0 + srow) * Kp + scs;  // N is a multiple of TBN

    // Issue async global->LDS DMAs for one stage (32B per thread per matrix).
    // IOFFSET is added to BOTH the LDS and the global address (ISA 15.18).
    auto stage_async = [&](int buf, int kt) {
        if (mOK) {
            unsigned   la = (unsigned)(uintptr_t)&As[buf][srow][scs];
            const bf16* ga = gA + kt;
            asm volatile("global_load_async_to_lds_b128 %0, %1, off"
                         :: "v"(la), "v"(ga) : "memory");
            asm volatile("global_load_async_to_lds_b128 %0, %1, off offset:16"
                         :: "v"(la), "v"(ga) : "memory");
        }
        {
            unsigned   lb = (unsigned)(uintptr_t)&Bs[buf][srow][scs];
            const bf16* gb = gB + kt;
            asm volatile("global_load_async_to_lds_b128 %0, %1, off"
                         :: "v"(lb), "v"(gb) : "memory");
            asm volatile("global_load_async_to_lds_b128 %0, %1, off offset:16"
                         :: "v"(lb), "v"(gb) : "memory");
        }
    };

    const int half = lane >> 4;
    const int l16  = lane & 15;
    const int kba  = half * 8;    // A: K-halves 0..15 / 16..31, 8-runs
    const int kbb  = half * 16;   // unused; B loads use kba (same k order as A)

    stage_async(0, 0);
    int cur = 0;
    for (int kt = 0; kt < Kp; kt += TBK) {
        // own asyncs done -> LDS writes visible; barrier -> everyone's visible
        asm volatile("s_wait_asynccnt 0" ::: "memory");
        __syncthreads();
        if (kt + TBK < Kp) stage_async(cur ^ 1, kt + TBK);   // overlap next fetch

        AFrag bfr[2];
#pragma unroll
        for (int j = 0; j < 2; ++j) {
            const int c = wn * 32 + j * 16 + l16;
            bfr[j].h[0] = *(const v8bf*)&Bs[cur][c][kba];
            bfr[j].h[1] = *(const v8bf*)&Bs[cur][c][16 + kba];
        }
#pragma unroll
        for (int i = 0; i < 4; ++i) {
            AFrag af;   // one A fragment live at a time -> no spills
            const int r = wm * 64 + i * 16 + l16;
            af.h[0] = *(const v8bf*)&As[cur][r][kba];
            af.h[1] = *(const v8bf*)&As[cur][r][16 + kba];
#pragma unroll
            for (int j = 0; j < 2; ++j)
                acc[i][j].v = __builtin_amdgcn_wmma_f32_16x16x32_bf16(
                    false, af.v, false, bfr[j].v,
                    (short)0, acc[i][j].v, false, false);
        }
        __syncthreads();   // all waves done reading 'cur' before it is refilled
        cur ^= 1;
    }

    // Epilogue: C/D layout -> VGPR r holds m = r + 8*(lane>>4), n = lane&15
    const int mh  = half * 8;
    const int phw = PH * PW;
#pragma unroll
    for (int i = 0; i < 4; ++i) {
#pragma unroll
        for (int j = 0; j < 2; ++j) {
            const int n = n0 + wn * 32 + j * 16 + l16;
            if (n >= N) continue;
            const int b  = n / phw;
            const int rm = n - b * phw;
            const int yy = rm / PW, xx = rm - yy * PW;
            const int oy = yy * sy + py, ox = xx * sx + px;
#pragma unroll
            for (int r = 0; r < 8; ++r) {
                const int m = m0 + wm * 64 + i * 16 + mh + r;
                if (m >= M) continue;
                const size_t idx = (((size_t)b * Cout + m) * OH + oy) * OW + ox;
                float v = acc[i][j].e[r];
                if (bias) v += bias[m];
                if (res)  v += res[idx];
                out[idx] = v;
            }
        }
    }
}

// ---------------------------------------------------------------------------
// im2col for standard conv: fp32 NCHW -> bf16 col[N][Kp] (k-contiguous),
// zero-pads k >= C*KH*KW. Consecutive gid = consecutive k -> coalesced writes,
// and consecutive kx -> adjacent input reads.
// ---------------------------------------------------------------------------
__global__ void im2col_bf16(const float* __restrict__ in, bf16* __restrict__ col,
                            int Bn, int C, int H, int W, int KH, int KW,
                            int S, int P, int OH, int OW, int Kp)
{
    const long long N     = (long long)Bn * OH * OW;
    const long long total = N * Kp;
    long long gid = (long long)blockIdx.x * blockDim.x + threadIdx.x;
    if (gid >= total) return;
    const int n = (int)(gid / Kp);
    const int k = (int)(gid - (long long)n * Kp);
    float val = 0.f;
    const int KK = C * KH * KW;
    if (k < KK) {
        const int c  = k / (KH * KW);
        const int rr = k - c * KH * KW;
        const int ky = rr / KW, kx = rr - ky * KW;
        const int b  = n / (OH * OW);
        const int sp = n - b * OH * OW;
        const int oy = sp / OW, ox = sp - oy * OW;
        const int iy = oy * S - P + ky, ix = ox * S - P + kx;
        if (iy >= 0 && iy < H && ix >= 0 && ix < W)
            val = in[(((size_t)b * C + c) * H + iy) * W + ix];
    }
    col[gid] = (bf16)val;
}

// ---------------------------------------------------------------------------
// im2col for ConvTranspose2d(k=4,s=2,p=1) stride-phase (py,px): Kp = C*4,
// col[N][Kp]. Output pixel (2a+py, 2c+px) reads input (a+py+ty-1, c+px+tx-1).
// ---------------------------------------------------------------------------
__global__ void im2colT_bf16(const float* __restrict__ in, bf16* __restrict__ col,
                             int Bn, int C, int H, int W, int py, int px)
{
    const int Kp = C * 4;
    const long long N     = (long long)Bn * H * W;
    const long long total = N * Kp;
    long long gid = (long long)blockIdx.x * blockDim.x + threadIdx.x;
    if (gid >= total) return;
    const int n = (int)(gid / Kp);
    const int k = (int)(gid - (long long)n * Kp);
    const int c  = k >> 2;
    const int t  = k & 3;
    const int ty = t >> 1, tx = t & 1;
    const int b  = n / (H * W);
    const int sp = n - b * H * W;
    const int a  = sp / W, cc = sp - a * W;
    const int iy = a + py + ty - 1;
    const int ix = cc + px + tx - 1;
    float val = 0.f;
    if (iy >= 0 && iy < H && ix >= 0 && ix < W)
        val = in[(((size_t)b * C + c) * H + iy) * W + ix];
    col[gid] = (bf16)val;
}

// OIHW fp32 -> bf16 [O][Kp] (Kp >= Kreal, zero-padded)
__global__ void pack_w_bf16(const float* __restrict__ w, bf16* __restrict__ out,
                            int O, int Kreal, int Kp)
{
    const long long total = (long long)O * Kp;
    long long gid = (long long)blockIdx.x * blockDim.x + threadIdx.x;
    if (gid >= total) return;
    const int o = (int)(gid / Kp);
    const int k = (int)(gid - (long long)o * Kp);
    out[gid] = (bf16)((k < Kreal) ? w[(size_t)o * Kreal + k] : 0.f);
}

// ConvT weight (Cin,Cout,4,4) -> phase-packed A[o][i*4 + ty*2 + tx] with flip
__global__ void pack_wT_phase_bf16(const float* __restrict__ w, bf16* __restrict__ out,
                                   int Cin, int Cout, int py, int px)
{
    const int total = Cout * Cin * 4;
    const int gid = blockIdx.x * blockDim.x + threadIdx.x;
    if (gid >= total) return;
    const int o = gid / (Cin * 4);
    const int r = gid - o * (Cin * 4);
    const int i = r >> 2;
    const int t = r & 3;
    const int ty = t >> 1, tx = t & 1;
    const int ky = py + 2 * ty, kx = px + 2 * tx;
    out[gid] = (bf16)w[(((size_t)i * Cout + o) * 4 + (3 - ky)) * 4 + (3 - kx)];
}

// Batch-norm training-mode stats per channel -> scale/shift
__global__ __launch_bounds__(256) void bn_stats(
    const float* __restrict__ x, const float* __restrict__ g, const float* __restrict__ b,
    float* __restrict__ scale, float* __restrict__ shift, int Bn, int C, int HW)
{
    const int c = blockIdx.x;
    const int total = Bn * HW;
    float s = 0.f, s2 = 0.f;
    for (int e = threadIdx.x; e < total; e += 256) {
        const int bi = e / HW, hw = e - bi * HW;
        const float v = x[((size_t)bi * C + c) * HW + hw];
        s += v; s2 += v * v;
    }
    __shared__ float sh[256], sh2[256];
    sh[threadIdx.x] = s; sh2[threadIdx.x] = s2;
    __syncthreads();
    for (int o = 128; o > 0; o >>= 1) {
        if (threadIdx.x < o) { sh[threadIdx.x] += sh[threadIdx.x + o];
                               sh2[threadIdx.x] += sh2[threadIdx.x + o]; }
        __syncthreads();
    }
    if (threadIdx.x == 0) {
        const float m   = sh[0] / total;
        const float var = sh2[0] / total - m * m;
        const float sc  = g[c] * rsqrtf(var + 1e-5f);
        scale[c] = sc;
        shift[c] = b[c] - m * sc;
    }
}

__global__ void bn_relu(const float* __restrict__ x, const float* __restrict__ scale,
                        const float* __restrict__ shift, float* __restrict__ y,
                        int C, int HW, long long total)
{
    long long gid = (long long)blockIdx.x * blockDim.x + threadIdx.x;
    if (gid >= total) return;
    const int c = (int)((gid / HW) % C);
    const float v = x[gid] * scale[c] + shift[c];
    y[gid] = v > 0.f ? v : 0.f;
}

// VQ: one block per pixel-vector; thread k scores code k; argmin w/ lowest-index tie
__global__ __launch_bounds__(128) void vq_argmin(
    const float* __restrict__ z, const float* __restrict__ cb,
    int* __restrict__ idx, int C, int HW, int Kc)
{
    const int n = blockIdx.x;
    const int b = n / HW, hw = n - b * HW;
    __shared__ float zs[256];
    for (int d = threadIdx.x; d < C; d += 128)
        zs[d] = z[((size_t)b * C + d) * HW + hw];
    __syncthreads();
    float s = 3.402823e38f;
    const int k = threadIdx.x;
    if (k < Kc) {
        s = 0.f;
        for (int d = 0; d < C; ++d) {
            const float cv = cb[(size_t)k * C + d];
            s += cv * cv - 2.f * cv * zs[d];   // ||c||^2 - 2 c.z (||z||^2 drops out)
        }
    }
    __shared__ float sv[128];
    __shared__ int   si[128];
    sv[threadIdx.x] = s; si[threadIdx.x] = k;
    __syncthreads();
    for (int o = 64; o > 0; o >>= 1) {
        if (threadIdx.x < o) {
            const float ov = sv[threadIdx.x + o];
            const int   oi = si[threadIdx.x + o];
            if (ov < sv[threadIdx.x] || (ov == sv[threadIdx.x] && oi < si[threadIdx.x])) {
                sv[threadIdx.x] = ov; si[threadIdx.x] = oi;
            }
        }
        __syncthreads();
    }
    if (threadIdx.x == 0) idx[n] = si[0];
}

// dec_in[b][c][hw] = codebook[idx[b*HW+hw]][c]
__global__ void vq_gather(const int* __restrict__ idx, const float* __restrict__ cb,
                          float* __restrict__ out, int C, int HW, long long total)
{
    long long gid = (long long)blockIdx.x * blockDim.x + threadIdx.x;
    if (gid >= total) return;
    const long long chw = (long long)C * HW;
    const int b  = (int)(gid / chw);
    const long long rm = gid - (long long)b * chw;
    const int c  = (int)(rm / HW);
    const int hw = (int)(rm - (long long)c * HW);
    out[gid] = cb[(size_t)idx[b * HW + hw] * C + c];
}

__global__ void zero1(float* p) { if (blockIdx.x == 0 && threadIdx.x == 0) *p = 0.f; }

__global__ __launch_bounds__(256) void vq_loss(
    const float* __restrict__ z, const float* __restrict__ e,
    float* __restrict__ loss, long long total, float coef)
{
    float s = 0.f;
    for (long long i = (long long)blockIdx.x * 256 + threadIdx.x; i < total;
         i += (long long)gridDim.x * 256) {
        const float d = z[i] - e[i];
        s += d * d;
    }
    __shared__ float sh[256];
    sh[threadIdx.x] = s;
    __syncthreads();
    for (int o = 128; o > 0; o >>= 1) {
        if (threadIdx.x < o) sh[threadIdx.x] += sh[threadIdx.x + o];
        __syncthreads();
    }
    if (threadIdx.x == 0) atomicAdd(loss, sh[0] * coef);
}

// ---------------------------------------------------------------------------
extern "C" void kernel_launch(void* const* d_in, const int* in_sizes, int n_in,
                              void* d_out, int out_size, void* d_ws, size_t ws_size,
                              hipStream_t stream)
{
    (void)in_sizes; (void)n_in; (void)out_size; (void)ws_size;

    const float* x        = (const float*)d_in[0];
    const float* enc_w1   = (const float*)d_in[1];
    const float* enc_b1   = (const float*)d_in[2];
    const float* enc_bn_g = (const float*)d_in[3];
    const float* enc_bn_b = (const float*)d_in[4];
    const float* enc_w2   = (const float*)d_in[5];
    const float* enc_b2   = (const float*)d_in[6];
    const float* rb_bn1_g = (const float*)d_in[7];
    const float* rb_bn1_b = (const float*)d_in[8];
    const float* rb_w3    = (const float*)d_in[9];
    const float* rb_b3    = (const float*)d_in[10];
    const float* rb_bn2_g = (const float*)d_in[11];
    const float* rb_bn2_b = (const float*)d_in[12];
    const float* rb_w1    = (const float*)d_in[13];
    const float* rb_b1    = (const float*)d_in[14];
    const float* dec_bn1_g = (const float*)d_in[15];
    const float* dec_bn1_b = (const float*)d_in[16];
    const float* dec_wt1   = (const float*)d_in[17];
    const float* dec_bt1   = (const float*)d_in[18];
    const float* dec_bn2_g = (const float*)d_in[19];
    const float* dec_bn2_b = (const float*)d_in[20];
    const float* dec_wt2   = (const float*)d_in[21];
    const float* dec_bt2   = (const float*)d_in[22];
    const float* codebook  = (const float*)d_in[23];

    // ---- workspace bump allocator ----
    char* wsb = (char*)d_ws;
    size_t off = 0;
    auto alloc = [&](size_t bytes) -> void* {
        void* p = wsb + off;
        off = (off + bytes + 255) & ~(size_t)255;
        return p;
    };

    bf16*  col  = (bf16*)alloc(134217728ULL * 2);      // shared im2col buffer (268 MB)
    float* q0   = (float*)alloc(33554432);             // 32x256x32x32 buffers
    float* q1   = (float*)alloc(33554432);
    float* q2   = (float*)alloc(33554432);
    float* q3   = (float*)alloc(33554432);
    float* q4   = (float*)alloc(33554432);
    float* u64a = (float*)alloc(134217728);            // 32x256x64x64 buffers
    float* u64b = (float*)alloc(134217728);
    bf16* pw_enc1 = (bf16*)alloc((size_t)256 * 64 * 2);
    bf16* pw_enc2 = (bf16*)alloc((size_t)256 * 4096 * 2);
    bf16* pw_rb3[4], *pw_rb1[4], *pw_dt1[4], *pw_dt2[4];
    for (int i = 0; i < 4; ++i) pw_rb3[i] = (bf16*)alloc((size_t)256 * 2304 * 2);
    for (int i = 0; i < 4; ++i) pw_rb1[i] = (bf16*)alloc((size_t)256 * 256 * 2);
    for (int i = 0; i < 4; ++i) pw_dt1[i] = (bf16*)alloc((size_t)256 * 1024 * 2);
    for (int i = 0; i < 4; ++i) pw_dt2[i] = (bf16*)alloc((size_t)3 * 1024 * 2);
    float* bnscale = (float*)alloc(256 * 4);
    float* bnshift = (float*)alloc(256 * 4);
    int*   vidx    = (int*)alloc(32768 * 4);

    auto gemm = [&](const bf16* W, const bf16* Bmat, const float* bias, const float* resi,
                    float* outp, int M, int N, int Kp, int Cout, int OH, int OW,
                    int PH, int PW, int sy, int sx, int py, int px) {
        dim3 grid(N / TBN, (M + TBM - 1) / TBM);
        gemm_bf16_wmma_nchw<<<grid, 256, 0, stream>>>(W, Bmat, bias, resi, outp,
            M, N, Kp, Cout, OH, OW, PH, PW, sy, sx, py, px);
    };
    auto im2col = [&](const float* in, int C, int H, int W, int KH, int KW,
                      int S, int P, int OH, int OW, int Kp) {
        long long total = (long long)Kp * 32 * OH * OW;
        im2col_bf16<<<(unsigned)((total + 255) / 256), 256, 0, stream>>>(
            in, col, 32, C, H, W, KH, KW, S, P, OH, OW, Kp);
    };
    auto im2colT = [&](const float* in, int C, int H, int W, int py, int px) {
        long long total = (long long)C * 4 * 32 * H * W;
        im2colT_bf16<<<(unsigned)((total + 255) / 256), 256, 0, stream>>>(
            in, col, 32, C, H, W, py, px);
    };
    auto bn = [&](const float* in, const float* g, const float* b, float* outp, int HW) {
        bn_stats<<<256, 256, 0, stream>>>(in, g, b, bnscale, bnshift, 32, 256, HW);
        long long total = 32LL * 256 * HW;
        bn_relu<<<(unsigned)((total + 255) / 256), 256, 0, stream>>>(
            in, bnscale, bnshift, outp, 256, HW, total);
    };
    auto resblock = [&](const float* in, float* outp, float* t1, float* t2, int ri) {
        bn(in, rb_bn1_g + ri * 256, rb_bn1_b + ri * 256, t1, 1024);
        im2col(t1, 256, 32, 32, 3, 3, 1, 1, 32, 32, 2304);
        gemm(pw_rb3[ri], col, rb_b3 + ri * 256, nullptr, t2,
             256, 32768, 2304, 256, 32, 32, 32, 32, 1, 1, 0, 0);
        bn(t2, rb_bn2_g + ri * 256, rb_bn2_b + ri * 256, t1, 1024);
        im2col(t1, 256, 32, 32, 1, 1, 1, 0, 32, 32, 256);
        gemm(pw_rb1[ri], col, rb_b1 + ri * 256, in, outp,
             256, 32768, 256, 256, 32, 32, 32, 32, 1, 1, 0, 0);
    };

    // ---- pack weights to bf16 ----
    pack_w_bf16<<<(256 * 64 + 255) / 256, 256, 0, stream>>>(enc_w1, pw_enc1, 256, 48, 64);
    pack_w_bf16<<<(256 * 4096) / 256, 256, 0, stream>>>(enc_w2, pw_enc2, 256, 4096, 4096);
    for (int i = 0; i < 4; ++i) {
        pack_w_bf16<<<(256 * 2304 + 255) / 256, 256, 0, stream>>>(
            rb_w3 + (size_t)i * 256 * 2304, pw_rb3[i], 256, 2304, 2304);
        pack_w_bf16<<<(256 * 256) / 256, 256, 0, stream>>>(
            rb_w1 + (size_t)i * 256 * 256, pw_rb1[i], 256, 256, 256);
    }
    for (int ph = 0; ph < 4; ++ph) {
        const int py = ph >> 1, px = ph & 1;
        pack_wT_phase_bf16<<<(256 * 256 * 4) / 256, 256, 0, stream>>>(
            dec_wt1, pw_dt1[ph], 256, 256, py, px);
        pack_wT_phase_bf16<<<(3 * 256 * 4 + 255) / 256, 256, 0, stream>>>(
            dec_wt2, pw_dt2[ph], 256, 3, py, px);
    }

    // ---- encoder ----
    im2col(x, 3, 128, 128, 4, 4, 2, 1, 64, 64, 64);
    gemm(pw_enc1, col, enc_b1, nullptr, u64a, 256, 131072, 64, 256, 64, 64, 64, 64, 1, 1, 0, 0);
    bn(u64a, enc_bn_g, enc_bn_b, u64b, 4096);
    im2col(u64b, 256, 64, 64, 4, 4, 2, 1, 32, 32, 4096);
    gemm(pw_enc2, col, enc_b2, nullptr, q0, 256, 32768, 4096, 256, 32, 32, 32, 32, 1, 1, 0, 0);
    resblock(q0, q3, q1, q2, 0);
    resblock(q3, q4, q1, q2, 1);                       // z = q4

    // ---- vector quantization + loss ----
    vq_argmin<<<32768, 128, 0, stream>>>(q4, codebook, vidx, 256, 1024, 128);
    {
        const long long total = 32LL * 256 * 1024;
        vq_gather<<<(unsigned)((total + 255) / 256), 256, 0, stream>>>(
            vidx, codebook, q0, 256, 1024, total);     // dec_in = embs (straight-through)
        float* lossp = ((float*)d_out) + 32LL * 3 * 128 * 128;
        zero1<<<1, 1, 0, stream>>>(lossp);
        vq_loss<<<1024, 256, 0, stream>>>(q4, q0, lossp, total, 1.5f / (float)total);
    }

    // ---- decoder ----
    resblock(q0, q3, q1, q2, 2);
    resblock(q3, q4, q1, q2, 3);
    bn(q4, dec_bn1_g, dec_bn1_b, q1, 1024);
    for (int ph = 0; ph < 4; ++ph) {                   // ConvT1: 4 stride-phases
        const int py = ph >> 1, px = ph & 1;
        im2colT(q1, 256, 32, 32, py, px);
        gemm(pw_dt1[ph], col, dec_bt1, nullptr, u64a,
             256, 32768, 1024, 256, 64, 64, 32, 32, 2, 2, py, px);
    }
    bn(u64a, dec_bn2_g, dec_bn2_b, u64b, 4096);
    for (int ph = 0; ph < 4; ++ph) {                   // ConvT2 -> x_pred in d_out
        const int py = ph >> 1, px = ph & 1;
        im2colT(u64b, 256, 64, 64, py, px);
        gemm(pw_dt2[ph], col, dec_bt2, nullptr, (float*)d_out,
             3, 131072, 1024, 3, 128, 128, 64, 64, 2, 2, py, px);
    }
}
